// BoardLoss_65189013619025
// MI455X (gfx1250) — compile-verified, matrix operand order checked by reading, not run on hardware
//
#include <hip/hip_runtime.h>

typedef int   v8i __attribute__((ext_vector_type(8)));
typedef float v4f __attribute__((ext_vector_type(4)));

#define BPT 8           // boards per thread; max hits = 12*8 = 96 <= 255 (fits u8 for IU8 WMMA)
#define BLOCK 256       // 8 waves of 32 on CDNA5

__global__ void bl_zero_ws(unsigned* __restrict__ ws) {
    if (threadIdx.x == 0 && blockIdx.x == 0) ws[0] = 0u;
}

__global__ __launch_bounds__(BLOCK)
void bl_main(const float* __restrict__ x, unsigned* __restrict__ ws,
             int batch, int total_threads) {
    const int tid = blockIdx.x * BLOCK + threadIdx.x;
    int acc = 0;

    const v4f* __restrict__ xb = (const v4f*)x;   // 36 floats = 9 x float4, 16B aligned (144B stride)

    #pragma unroll
    for (int k = 0; k < BPT; ++k) {
        const int board = tid + k * total_threads;
        if (board < batch) {
            const v4f* p = xb + (size_t)board * 9;
            unsigned long long m = 0ull;          // bit (r*6+c) = (x[r][c] > 0.5)
            #pragma unroll
            for (int i = 0; i < 9; ++i) {
                v4f q = __builtin_nontemporal_load(p + i);   // global_load_b128 th:NT
                const int base = i * 4;
                if (q.x > 0.5f) m |= (1ull << (base + 0));
                if (q.y > 0.5f) m |= (1ull << (base + 1));
                if (q.z > 0.5f) m |= (1ull << (base + 2));
                if (q.w > 0.5f) m |= (1ull << (base + 3));
            }
            // ---- horizontal triples (all 6 rows at once) ----
            // eq of neighbors valid at c=0..4 -> 0x1F per row; triples at c=0..3 -> 0xF per row
            unsigned long long e = ~(m ^ (m >> 1)) & 0x7DF7DF7DFull;
            unsigned long long t =  e & (e >> 1)   & 0x3CF3CF3CFull;
            unsigned long long g = t | (t >> 1); g |= (g >> 2);      // OR 4 trip bits into bit 6r
            acc += __popcll(g & 0x41041041ull);                      // row hits
            // ---- vertical triples (all 6 cols at once) ----
            unsigned long long e6 = ~(m ^ (m >> 6)) & 0x3FFFFFFFull; // r=0..4
            unsigned long long t6 =  e6 & (e6 >> 6) & 0xFFFFFFull;   // r=0..3
            unsigned long long g6 = t6 | (t6 >> 6); g6 |= (g6 >> 12);
            acc += __popcll(g6 & 0x3Full);                           // col hits
        }
    }

    // ---- exact wave-level integer reduction via V_WMMA_I32_16X16X64_IU8 ----
    // Each lane contributes its count (<=96, fits u8) as one A element; B = all ones.
    // D[m][n] = sum_k A[m][k]; summing the 8 D VGPRs gives lanes 0..15 the sum of
    // rows 0..7 and lanes 16..31 the sum of rows 8..15 -> lane0 + lane16 = wave total.
    v8i a = {0,0,0,0,0,0,0,0};
    a[0] = acc;
    v8i b = {0x01010101,0x01010101,0x01010101,0x01010101,
             0x01010101,0x01010101,0x01010101,0x01010101};
    v8i c = {0,0,0,0,0,0,0,0};
    v8i d = __builtin_amdgcn_wmma_i32_16x16x64_iu8(false, a, false, b, c, false, false);
    int s = d[0] + d[1] + d[2] + d[3] + d[4] + d[5] + d[6] + d[7];
    int wave_total = __builtin_amdgcn_readlane(s, 0) + __builtin_amdgcn_readlane(s, 16);

    if ((threadIdx.x & 31u) == 0u)
        atomicAdd(ws, (unsigned)wave_total);     // integer atomic: bit-exact, order-free
}

__global__ void bl_finalize(const unsigned* __restrict__ ws, float* __restrict__ out, float inv) {
    if (threadIdx.x == 0 && blockIdx.x == 0) out[0] = (float)ws[0] * inv;
}

extern "C" void kernel_launch(void* const* d_in, const int* in_sizes, int n_in,
                              void* d_out, int out_size, void* d_ws, size_t ws_size,
                              hipStream_t stream) {
    const float* x = (const float*)d_in[0];
    float* out = (float*)d_out;
    unsigned* ws = (unsigned*)d_ws;

    const int batch = in_sizes[0] / 36;                       // 2,000,000
    const int threads_needed = (batch + BPT - 1) / BPT;
    const int blocks = (threads_needed + BLOCK - 1) / BLOCK;
    const int total_threads = blocks * BLOCK;
    const float inv = 1.0f / (6.0f * (float)batch);

    bl_zero_ws<<<1, 1, 0, stream>>>(ws);
    bl_main<<<blocks, BLOCK, 0, stream>>>(x, ws, batch, total_threads);
    bl_finalize<<<1, 1, 0, stream>>>(ws, out, inv);
}
